// GraphTransformer_84997402788496
// MI455X (gfx1250) — matrix operand order repaired and checked by reading, not compiled, not run on hardware
//
#include <hip/hip_runtime.h>
#include <stdint.h>
#include <stddef.h>

typedef __bf16 bf16;
typedef __attribute__((ext_vector_type(16))) __bf16 v16bf;
typedef __attribute__((ext_vector_type(8)))  float  v8f;
typedef __attribute__((ext_vector_type(4)))  int    v4i;
typedef __attribute__((address_space(3))) v4i* lds_v4i_ptr;

constexpr int NB = 8, NS = 512, ND = 768, NH = 12, NDR = 64, NDFF = 1024, NV = 150;
constexpr int NM = NB * NS;   // 4096 token rows

#if defined(__gfx1250__) && __has_builtin(__builtin_amdgcn_global_load_async_to_lds_b128) && __has_builtin(__builtin_amdgcn_s_wait_asynccnt)
#define HAVE_ASYNC_LDS 1
#else
#define HAVE_ASYNC_LDS 0
#endif

__device__ __forceinline__ v8f wmma_bf16(v16bf a, v16bf b, v8f c) {
  // D(f32,16x16) = A(bf16,16x32) * B(bf16,32x16) + C
  return __builtin_amdgcn_wmma_f32_16x16x32_bf16(false, a, false, b, (short)0, c, false, false);
}

// A-operand (16x32 bf16): lane l -> M=l%16, half=l/16.
// elements 0..7  hold K =  8*half + 0..7
// elements 8..15 hold K = 16 + 8*half + 0..7
__device__ __forceinline__ v16bf frag_a(const bf16* rowptr, int half) {
  v16bf r;
#pragma unroll
  for (int i = 0; i < 8; ++i) r[i] = rowptr[8 * half + i];
#pragma unroll
  for (int i = 0; i < 8; ++i) r[8 + i] = rowptr[16 + 8 * half + i];
  return r;
}
// B-operand (32x16 bf16): lane l -> N=l%16, half=l/16; elements 0..7 hold Kdim = 8*half + 0..7, 8..15 hold Kdim = 16 + 8*half + 0..7
// rowptr must point at the K-major row for this lane's N column (contiguous 32B).
__device__ __forceinline__ v16bf frag_b(const bf16* rowptr, int half) {
  v16bf r;
#pragma unroll
  for (int i = 0; i < 8; ++i) r[i] = rowptr[8 * half + i];
#pragma unroll
  for (int i = 0; i < 8; ++i) r[8 + i] = rowptr[16 + 8 * half + i];
  return r;
}

// ---------------------------------------------------------------------------
// x = src_emb + sinusoidal positional encoding
// ---------------------------------------------------------------------------
__global__ __launch_bounds__(256) void k_add_pos(const float* __restrict__ src, float* __restrict__ x) {
  int i = blockIdx.x * 256 + threadIdx.x;
  if (i >= NB * NS * ND) return;
  int d = i % ND;
  int s = (i / ND) % NS;
  int ih = d >> 1;
  float f = __expf(-(float)(2 * ih) * (9.210340371976184f / (float)ND)); // ln(10000)
  float ang = (float)s * f;
  float pe = (d & 1) ? __cosf(ang) : __sinf(ang);
  x[i] = src[i] + pe;
}

// ---------------------------------------------------------------------------
// Row LayerNorm over D=768. One 256-thread block per row (3 elems/thread).
// Writes bf16 (GEMM input) and/or f32.
// ---------------------------------------------------------------------------
__global__ __launch_bounds__(256) void k_layernorm(const float* __restrict__ x,
                                                   const float* __restrict__ g,
                                                   const float* __restrict__ bta,
                                                   bf16* __restrict__ outB,
                                                   float* __restrict__ outF) {
  __shared__ float red[16];
  const int row = blockIdx.x, tid = threadIdx.x;
  const float* xr = x + (size_t)row * ND;
  float v0 = xr[tid], v1 = xr[tid + 256], v2 = xr[tid + 512];
  float s = v0 + v1 + v2;
  float s2 = v0 * v0 + v1 * v1 + v2 * v2;
#pragma unroll
  for (int o = 1; o < 32; o <<= 1) { s += __shfl_xor(s, o, 32); s2 += __shfl_xor(s2, o, 32); }
  int w = tid >> 5;
  if ((tid & 31) == 0) { red[w] = s; red[8 + w] = s2; }
  __syncthreads();
  if (tid == 0) {
    float a = 0.f, b2 = 0.f;
#pragma unroll
    for (int i = 0; i < 8; ++i) { a += red[i]; b2 += red[8 + i]; }
    red[0] = a; red[8] = b2;
  }
  __syncthreads();
  float mean = red[0] * (1.0f / ND);
  float var  = red[8] * (1.0f / ND) - mean * mean;
  float rstd = rsqrtf(var + 1e-5f);
#pragma unroll
  for (int t = 0; t < 3; ++t) {
    int c = tid + t * 256;
    float val = (xr[c] - mean) * rstd * g[c] + bta[c];
    size_t oi = (size_t)row * ND + c;
    if (outB) outB[oi] = (bf16)val;
    if (outF) outF[oi] = val;
  }
}

// ---------------------------------------------------------------------------
__global__ __launch_bounds__(256) void k_f32_to_bf16(const float* __restrict__ in,
                                                     bf16* __restrict__ out, int n) {
  int i = blockIdx.x * 256 + threadIdx.x;
  if (i < n) out[i] = (bf16)in[i];
}

// ---------------------------------------------------------------------------
// Tiled WMMA GEMM: out[M,N] = A(bf16)[M,K] @ W(bf16)[K,N] + bias (+res) (relu)
// 128 threads = 4 waves; block tile 64x64; each wave 32x32 (2x2 WMMA); K-step 32.
// ---------------------------------------------------------------------------
__global__ __launch_bounds__(128) void k_gemm(const bf16* __restrict__ A, const bf16* __restrict__ W,
                                              const float* __restrict__ bias, const float* __restrict__ res,
                                              float* __restrict__ outF, bf16* __restrict__ outB,
                                              int M, int N, int K, int relu, float oscale) {
  __shared__ bf16 As[64][32];    // [m][k]
  __shared__ bf16 BsT[64][32];   // [n][k]  (transposed during staging)
  const int tid = threadIdx.x;
  const int wave = tid >> 5, lane = tid & 31;
  const int half = lane >> 4, lN = lane & 15;
  const int m0 = blockIdx.y * 64, n0 = blockIdx.x * 64;
  const int wr = wave >> 1, wc = wave & 1;

  v8f acc[2][2];
#pragma unroll
  for (int i = 0; i < 2; ++i)
#pragma unroll
    for (int j = 0; j < 2; ++j) acc[i][j] = v8f{0.f,0.f,0.f,0.f,0.f,0.f,0.f,0.f};

  const int ar = tid >> 1, ac = (tid & 1) * 16;       // A stage: 16 elems/thread
  const int wk = tid >> 2, wcseg = (tid & 3) * 16;    // W stage: 16 elems/thread

  for (int k0 = 0; k0 < K; k0 += 32) {
    const bf16* asrc = A + (size_t)(m0 + ar) * K + k0 + ac;
#if HAVE_ASYNC_LDS
    // toolchain signature: (v4i* src_global, as(3) v4i* dst_lds, imm offset, imm cpol)
    __builtin_amdgcn_global_load_async_to_lds_b128(
        (v4i*)asrc, (lds_v4i_ptr)&As[ar][ac], 0, 0);
    __builtin_amdgcn_global_load_async_to_lds_b128(
        (v4i*)(asrc + 8), (lds_v4i_ptr)&As[ar][ac + 8], 0, 0);
#else
    ((uint4*)&As[ar][ac])[0] = ((const uint4*)asrc)[0];
    ((uint4*)&As[ar][ac])[1] = ((const uint4*)asrc)[1];
#endif
    {
      const bf16* wsrc = W + (size_t)(k0 + wk) * N + n0 + wcseg;
      __align__(16) bf16 tmp[16];
      ((uint4*)tmp)[0] = ((const uint4*)wsrc)[0];
      ((uint4*)tmp)[1] = ((const uint4*)wsrc)[1];
#pragma unroll
      for (int j = 0; j < 16; ++j) BsT[wcseg + j][wk] = tmp[j];
    }
    if (k0 + 32 < K) {  // global_prefetch_b8 of next K tiles
      __builtin_prefetch(A + (size_t)(m0 + ar) * K + k0 + 32 + ac, 0, 1);
      __builtin_prefetch(W + (size_t)(k0 + 32 + wk) * N + n0 + wcseg, 0, 1);
    }
#if HAVE_ASYNC_LDS
    __builtin_amdgcn_s_wait_asynccnt(0);
#endif
    __syncthreads();

#pragma unroll
    for (int i = 0; i < 2; ++i) {
      v16bf a = frag_a(&As[32 * wr + 16 * i + lN][0], half);
#pragma unroll
      for (int j = 0; j < 2; ++j) {
        v16bf b = frag_b(&BsT[32 * wc + 16 * j + lN][0], half);
        acc[i][j] = wmma_bf16(a, b, acc[i][j]);
      }
    }
    __syncthreads();
  }

#pragma unroll
  for (int i = 0; i < 2; ++i)
#pragma unroll
    for (int j = 0; j < 2; ++j)
#pragma unroll
      for (int r = 0; r < 8; ++r) {
        int grow = m0 + 32 * wr + 16 * i + r + 8 * half;   // C layout: M = r + 8*(lane/16)
        int gcol = n0 + 32 * wc + 16 * j + lN;
        float v = acc[i][j][r] + bias[gcol];
        if (relu) v = fmaxf(v, 0.f);
        size_t oi = (size_t)grow * N + gcol;
        if (res) v += res[oi];
        if (outF) outF[oi] = v;
        if (outB) outB[oi] = (bf16)(v * oscale);
      }
}

// ---------------------------------------------------------------------------
// Flash attention with relation-embedding structure bias & structure context.
// One wave per (b, h, 16-query tile). rel_emb (38.4KB) cached in LDS; the
// 512MB structure tensor is never materialized. Content QK^T and P@V use
// WMMA bf16; per-q-row structure terms use LDS gathers (VALU FMA).
// q is pre-scaled by 1/sqrt(64).
// ---------------------------------------------------------------------------
__global__ __launch_bounds__(32) void k_attention(const bf16* __restrict__ qb, const bf16* __restrict__ kb,
                                                  const bf16* __restrict__ vb, const int* __restrict__ rel,
                                                  const float* __restrict__ rel_emb,
                                                  const int* __restrict__ src_len,
                                                  bf16* __restrict__ ctx) {
  __shared__ float emb_lds[NV * NDR];   // 38400 B
  __shared__ float q_lds[16 * NDR];     // scaled q tile, f32
  __shared__ float P_lds[16 * 32];      // probabilities of current 32-key block
  __shared__ float scale_lds[16];       // per-row online-softmax rescale
  __shared__ float l_lds[16];           // per-row denominators
  __shared__ bf16  VT[NDR * 32];        // V^T tile [d][k]
  __shared__ float ctx_lds[16 * NDR];

  const int lane = threadIdx.x;
  const int b = blockIdx.z, h = blockIdx.y;
  const int q0 = blockIdx.x * 16;
  const int len = src_len[b];
  const int half = lane >> 4, lN = lane & 15, Mb = 8 * half;

  for (int i = lane; i < NV * NDR; i += 32) emb_lds[i] = rel_emb[i];
  for (int i = lane; i < 16 * NDR; i += 32) {
    int r = i >> 6, d = i & 63;
    q_lds[i] = (float)qb[((size_t)(b * NS + q0 + r)) * ND + h * NDR + d];
  }
  __syncthreads();

  const bf16* qrow = qb + ((size_t)(b * NS + q0 + lN)) * ND + h * NDR;
  v16bf a0 = frag_a(qrow, half);        // d 0..31
  v16bf a1 = frag_a(qrow + 32, half);   // d 32..63

  v8f acc0 = v8f{0,0,0,0,0,0,0,0}, acc1 = acc0, acc2 = acc0, acc3 = acc0;  // content ctx, d 0..63
  float cs[32];                                                            // structure ctx slice
#pragma unroll
  for (int j = 0; j < 32; ++j) cs[j] = 0.f;
  float mrow[8], lrow[8];
#pragma unroll
  for (int r = 0; r < 8; ++r) { mrow[r] = -3.0e38f; lrow[r] = 0.f; }

  for (int kt = 0; kt < NS / 32; ++kt) {
    const int k0 = kt * 32;
    // ---- content scores: two 16x16 C tiles (keys k0..15 / k0+16..31) -------
    const bf16* krl = kb + ((size_t)(b * NS + k0 + lN)) * ND + h * NDR;
    const bf16* krh = krl + (size_t)16 * ND;
    v8f clo = v8f{0,0,0,0,0,0,0,0}, chi = clo;
    clo = wmma_bf16(a0, frag_b(krl, half), clo);
    clo = wmma_bf16(a1, frag_b(krl + 32, half), clo);
    chi = wmma_bf16(a0, frag_b(krh, half), chi);
    chi = wmma_bf16(a1, frag_b(krh + 32, half), chi);

    // ---- structure bias, mask, online softmax ------------------------------
    const int* relq = rel + ((size_t)b * NS + q0) * NS;
#pragma unroll
    for (int r = 0; r < 8; ++r) {
      const int M = Mb + r;
      const int* rr = relq + (size_t)M * NS + k0;
      const int ilo = rr[lN], ihi = rr[16 + lN];
      const float* elo = emb_lds + ilo * NDR;
      const float* ehi = emb_lds + ihi * NDR;
      const float* qr  = q_lds + M * NDR;
      float s0 = 0.f, s1 = 0.f;
      for (int d = 0; d < NDR; ++d) { float qv = qr[d]; s0 += qv * elo[d]; s1 += qv * ehi[d]; }
      float vlo = clo[r] + s0, vhi = chi[r] + s1;
      if (k0 + lN >= len)      vlo = -1e9f;
      if (k0 + 16 + lN >= len) vhi = -1e9f;

      float mx = fmaxf(vlo, vhi);
#pragma unroll
      for (int o = 1; o < 16; o <<= 1) mx = fmaxf(mx, __shfl_xor(mx, o, 32));
      float mnew = fmaxf(mrow[r], mx);
      float sc = __expf(mrow[r] - mnew);
      float plo = __expf(vlo - mnew), phi = __expf(vhi - mnew);
      float ps = plo + phi;
#pragma unroll
      for (int o = 1; o < 16; o <<= 1) ps += __shfl_xor(ps, o, 32);
      lrow[r] = lrow[r] * sc + ps;
      mrow[r] = mnew;
      acc0[r] *= sc; acc1[r] *= sc; acc2[r] *= sc; acc3[r] *= sc;
      P_lds[M * 32 + lN] = plo;
      P_lds[M * 32 + 16 + lN] = phi;
      if (lN == 0) scale_lds[M] = sc;
    }
    __syncthreads();

    // ---- stage V^T tile (lane = one key row) -------------------------------
    {
      const bf16* vr = vb + ((size_t)(b * NS + k0 + lane)) * ND + h * NDR;
      for (int d = 0; d < NDR; ++d) VT[d * 32 + lane] = vr[d];
    }
    __syncthreads();

    // ---- content context: P (16x32) @ V (32x64) via 4 WMMAs ----------------
    v16bf pa;
#pragma unroll
    for (int e = 0; e < 8; ++e)  pa[e] = (bf16)P_lds[lN * 32 + (e + 8 * half)];
#pragma unroll
    for (int e = 8; e < 16; ++e) pa[e] = (bf16)P_lds[lN * 32 + (e + 8 + 8 * half)];
    acc0 = wmma_bf16(pa, frag_b(&VT[(0  + lN) * 32], half), acc0);
    acc1 = wmma_bf16(pa, frag_b(&VT[(16 + lN) * 32], half), acc1);
    acc2 = wmma_bf16(pa, frag_b(&VT[(32 + lN) * 32], half), acc2);
    acc3 = wmma_bf16(pa, frag_b(&VT[(48 + lN) * 32], half), acc3);

    // ---- structure context: per-row gather-weighted sum --------------------
    {
      const int row = lN, dh = half;   // lane owns (row, 32*dh..32*dh+31)
      const float scr = scale_lds[row];
#pragma unroll
      for (int j = 0; j < 32; ++j) cs[j] *= scr;
      const int* rr = rel + ((size_t)b * NS + q0 + row) * NS + k0;
      for (int k = 0; k < 32; ++k) {
        float p = P_lds[row * 32 + k];
        const float* e = emb_lds + rr[k] * NDR + 32 * dh;
        for (int j = 0; j < 32; ++j) cs[j] += p * e[j];
      }
    }
    __syncthreads();
  }

  // ---- finalize: combine layouts through LDS, divide by l, write bf16 ------
  if (lN == 0) {
#pragma unroll
    for (int r = 0; r < 8; ++r) l_lds[Mb + r] = lrow[r];
  }
  for (int i = lane; i < 16 * NDR; i += 32) ctx_lds[i] = 0.f;
  __syncthreads();
#pragma unroll
  for (int r = 0; r < 8; ++r) {
    const int M = Mb + r;
    ctx_lds[M * NDR + 0  + lN] = acc0[r];
    ctx_lds[M * NDR + 16 + lN] = acc1[r];
    ctx_lds[M * NDR + 32 + lN] = acc2[r];
    ctx_lds[M * NDR + 48 + lN] = acc3[r];
  }
  __syncthreads();
  {
    const int row = lN, dh = half;
    const float invl = 1.0f / l_lds[row];
    bf16* op = ctx + ((size_t)(b * NS + q0 + row)) * ND + h * NDR + 32 * dh;
    for (int j = 0; j < 32; ++j)
      op[j] = (bf16)((ctx_lds[row * NDR + 32 * dh + j] + cs[j]) * invl);
  }
}

// ---------------------------------------------------------------------------
// d_in flattening (JAX pytree, dict keys sorted):
//  0 src_emb, 1 src_length, 2 relations,
//  per layer l (base 3+16*l): k.b,k.w,ln1_b,ln1_g,ln2_b,ln2_g,o.b,o.w,q.b,q.w,
//                             v.b,v.w,w1.b,w1.w,w2.b,w2.w
//  35 lnf_b, 36 lnf_g, 37 rel_emb
// ---------------------------------------------------------------------------
extern "C" void kernel_launch(void* const* d_in, const int* in_sizes, int n_in,
                              void* d_out, int out_size, void* d_ws, size_t ws_size,
                              hipStream_t stream) {
  (void)in_sizes; (void)n_in; (void)out_size; (void)ws_size;
  const float* src_emb = (const float*)d_in[0];
  const int*   src_len = (const int*)d_in[1];
  const int*   rel     = (const int*)d_in[2];
  const float* lnf_b   = (const float*)d_in[35];
  const float* lnf_g   = (const float*)d_in[36];
  const float* rel_emb = (const float*)d_in[37];

  char* base = (char*)d_ws;
  float* X  = (float*)(base + 0);          // 12,582,912 B  activations f32
  bf16*  HB = (bf16*)(base + 12582912);    //  6,291,456 B  LN output bf16
  bf16*  WB = (bf16*)(base + 18874368);    //  1,572,864 B  current weight bf16
  bf16*  QB = (bf16*)(base + 20447232);    //  6,291,456 B
  bf16*  KB = (bf16*)(base + 26738688);    //  6,291,456 B
  bf16*  VB = (bf16*)(base + 33030144);    //  6,291,456 B
  bf16*  CX = (bf16*)(base + 39321600);    //  6,291,456 B  attention context
  bf16*  F1 = (bf16*)(base + 20447232);    //  8,388,608 B  (aliases QB/KB, dead post-attn)

  const dim3 b256(256), b128(128), b32(32);
  k_add_pos<<<dim3((NB * NS * ND + 255) / 256), b256, 0, stream>>>(src_emb, X);

  for (int l = 0; l < 2; ++l) {
    const int bi = 3 + 16 * l;
    const float* k_b  = (const float*)d_in[bi + 0];
    const float* k_w  = (const float*)d_in[bi + 1];
    const float* ln1b = (const float*)d_in[bi + 2];
    const float* ln1g = (const float*)d_in[bi + 3];
    const float* ln2b = (const float*)d_in[bi + 4];
    const float* ln2g = (const float*)d_in[bi + 5];
    const float* o_b  = (const float*)d_in[bi + 6];
    const float* o_w  = (const float*)d_in[bi + 7];
    const float* q_b  = (const float*)d_in[bi + 8];
    const float* q_w  = (const float*)d_in[bi + 9];
    const float* v_b  = (const float*)d_in[bi + 10];
    const float* v_w  = (const float*)d_in[bi + 11];
    const float* w1_b = (const float*)d_in[bi + 12];
    const float* w1_w = (const float*)d_in[bi + 13];
    const float* w2_b = (const float*)d_in[bi + 14];
    const float* w2_w = (const float*)d_in[bi + 15];

    k_layernorm<<<NM, b256, 0, stream>>>(X, ln1g, ln1b, HB, (float*)nullptr);

    // Q (scaled by 1/sqrt(64)), K, V projections -> bf16 [B,S,H*DR]
    k_f32_to_bf16<<<(ND * ND + 255) / 256, b256, 0, stream>>>(q_w, WB, ND * ND);
    k_gemm<<<dim3(ND / 64, NM / 64), b128, 0, stream>>>(HB, WB, q_b, nullptr, nullptr, QB,
                                                        NM, ND, ND, 0, 0.125f);
    k_f32_to_bf16<<<(ND * ND + 255) / 256, b256, 0, stream>>>(k_w, WB, ND * ND);
    k_gemm<<<dim3(ND / 64, NM / 64), b128, 0, stream>>>(HB, WB, k_b, nullptr, nullptr, KB,
                                                        NM, ND, ND, 0, 1.0f);
    k_f32_to_bf16<<<(ND * ND + 255) / 256, b256, 0, stream>>>(v_w, WB, ND * ND);
    k_gemm<<<dim3(ND / 64, NM / 64), b128, 0, stream>>>(HB, WB, v_b, nullptr, nullptr, VB,
                                                        NM, ND, ND, 0, 1.0f);

    k_attention<<<dim3(NS / 16, NH, NB), b32, 0, stream>>>(QB, KB, VB, rel, rel_emb, src_len, CX);

    // output projection + residual into X
    k_f32_to_bf16<<<(ND * ND + 255) / 256, b256, 0, stream>>>(o_w, WB, ND * ND);
    k_gemm<<<dim3(ND / 64, NM / 64), b128, 0, stream>>>(CX, WB, o_b, X, X, nullptr,
                                                        NM, ND, ND, 0, 1.0f);

    // FFN
    k_layernorm<<<NM, b256, 0, stream>>>(X, ln2g, ln2b, HB, (float*)nullptr);
    k_f32_to_bf16<<<(ND * NDFF + 255) / 256, b256, 0, stream>>>(w1_w, WB, ND * NDFF);
    k_gemm<<<dim3(NDFF / 64, NM / 64), b128, 0, stream>>>(HB, WB, w1_b, nullptr, nullptr, F1,
                                                          NM, NDFF, ND, 1, 1.0f);
    k_f32_to_bf16<<<(NDFF * ND + 255) / 256, b256, 0, stream>>>(w2_w, WB, NDFF * ND);
    k_gemm<<<dim3(ND / 64, NM / 64), b128, 0, stream>>>(F1, WB, w2_b, X, X, nullptr,
                                                        NM, ND, NDFF, 0, 1.0f);
  }

  k_layernorm<<<NM, b256, 0, stream>>>(X, lnf_g, lnf_b, (bf16*)nullptr, (float*)d_out);
}
